// HeteroGraphTransLayer_1958505087672
// MI455X (gfx1250) — matrix operand order repaired and checked by reading, not compiled, not run on hardware
//
#include <hip/hip_runtime.h>

typedef __attribute__((ext_vector_type(16))) _Float16 v16h;
typedef __attribute__((ext_vector_type(8)))  _Float16 v8h;
typedef __attribute__((ext_vector_type(8)))  float    v8f;

#define NN   50000
#define NE   800000
#define DIN  128
#define DHID 128
#define NH   8
#define DKH  16
#define TNT  8
#define TET  8
#define NB   4

// ---------------- workspace layout (bytes, all naturally 256B aligned) ----------------
// Wt:    [4 proj][8 type][out=128][in=128] f16 (B-matrix, transposed so K is contiguous)
constexpr size_t OFF_WT    = 0;
constexpr size_t OFF_LOOPW = OFF_WT    + 4ull * TNT * DHID * DIN * 2;   // [out][in] f16
constexpr size_t OFF_NODEF = OFF_LOOPW + (size_t)DHID * DIN * 2;        // node_feat f16 [N][128]
constexpr size_t OFF_KF    = OFF_NODEF + (size_t)NN * DIN  * 2;         // k f16 [N][128]
constexpr size_t OFF_QF    = OFF_KF    + (size_t)NN * DHID * 2;         // q f16
constexpr size_t OFF_VF    = OFF_QF    + (size_t)NN * DHID * 2;         // v f16
constexpr size_t OFF_AGGH  = OFF_VF    + (size_t)NN * DHID * 2;         // agg f16 [N][128]
constexpr size_t OFF_EB    = OFF_AGGH  + (size_t)NN * DHID * 2;         // e / exp(e) f32 [E][8]
constexpr size_t OFF_MB    = OFF_EB    + (size_t)NE * NH * 4;           // seg-max keys u32 [N][8]
constexpr size_t OFF_DB    = OFF_MB    + (size_t)NN * NH * 4;           // seg-sum f32 [N][8]
constexpr size_t OFF_AGGF  = OFF_DB    + (size_t)NN * NH * 4;           // agg f32 [N][128]

// order-preserving float<->uint key for atomicMax-based segment max
__device__ __forceinline__ unsigned fkey(float f) {
  unsigned u = __float_as_uint(f);
  return (u & 0x80000000u) ? ~u : (u | 0x80000000u);
}
__device__ __forceinline__ float fdekey(unsigned k) {
  unsigned u = (k & 0x80000000u) ? (k & 0x7FFFFFFFu) : ~k;
  return __uint_as_float(u);
}

// ---------------- K0: combine bases -> per-type weights, transposed f16 ----------------
__global__ __launch_bounds__(256) void hgt_prep_w(
    const float* ck, const float* bk, const float* cq, const float* bq,
    const float* cv, const float* bv, const float* ca, const float* ba,
    const float* loopw, _Float16* wt, _Float16* loopwt) {
  int idx = blockIdx.x * 256 + threadIdx.x;
  const int PER_P = TNT * DHID * DIN;           // 131072
  if (idx < 4 * PER_P) {
    int p = idx / PER_P;
    int r = idx % PER_P;
    int t = r / (DHID * DIN);
    int r2 = r % (DHID * DIN);
    int o = r2 / DIN;                            // output col
    int i = r2 % DIN;                            // input (K) index
    const float* comp  = (p == 0) ? ck : (p == 1) ? cq : (p == 2) ? cv : ca;
    const float* basis = (p == 0) ? bk : (p == 1) ? bq : (p == 2) ? bv : ba;
    float s = 0.f;
#pragma unroll
    for (int b = 0; b < NB; ++b)
      s += comp[t * NB + b] * basis[(size_t)b * (DIN * DHID) + i * DHID + o];
    wt[idx] = (_Float16)s;                       // idx ordering == [p][t][o][i]
  } else if (idx < 4 * PER_P + DHID * DIN) {
    int j = idx - 4 * PER_P;
    int o = j / DIN, i = j % DIN;
    loopwt[j] = (_Float16)loopw[i * DHID + o];   // transpose
  }
}

// ---------------- K1: node features -> f16 ----------------
__global__ __launch_bounds__(256) void hgt_cvt_nodes(const float* nf, _Float16* nh_) {
  int idx = blockIdx.x * 256 + threadIdx.x;
  if (idx < NN * DIN) nh_[idx] = (_Float16)nf[idx];
}

// ---------------- K2: init accumulators ----------------
__global__ __launch_bounds__(256) void hgt_init(float* aggf, unsigned* mb, float* db) {
  int idx = blockIdx.x * 256 + threadIdx.x;
  if (idx < NN * DHID) aggf[idx] = 0.f;
  if (idx < NN * NH) { mb[idx] = 0u; db[idx] = 0.f; }  // key 0 < key(-inf)
}

// ---------------- K3: k/q/v projections via masked-type WMMA ----------------
// grid: (3125 node tiles, 3 projections); block 256 = 8 waves (one 16-col tile each)
__global__ __launch_bounds__(256) void hgt_proj(
    const _Float16* __restrict__ nodef, const int* __restrict__ ntype,
    const _Float16* __restrict__ wt,
    _Float16* __restrict__ kf, _Float16* __restrict__ qf, _Float16* __restrict__ vf) {
  const int tile = blockIdx.x;
  const int p    = blockIdx.y;
  const int wave = threadIdx.x >> 5;
  const int lane = threadIdx.x & 31;
  const int half = lane >> 4, lrow = lane & 15;
  const int rowA = tile * 16 + lrow;             // A-row owned by this lane (both halves)
  const int ty   = ntype[rowA];
  const int col  = wave * 16 + lrow;             // B/D column owned by this lane
  const _Float16* wbase = wt + (size_t)p * (TNT * DHID * DIN);
  const v16h vz = {};
  v8f c = {};
#pragma unroll
  for (int kk = 0; kk < 4; ++kk) {
    union { v16h v; v8h h[2]; } a;
    const _Float16* pa = nodef + (size_t)rowA * DIN + kk * 32 + half * 16;
    a.h[0] = *(const v8h*)(pa);
    a.h[1] = *(const v8h*)(pa + 8);
#pragma unroll
    for (int t = 0; t < TNT; ++t) {
      v16h b = *(const v16h*)(wbase + ((size_t)t * DHID + col) * DIN + kk * 32 + half * 16);
      v16h at = (ty == t) ? a.v : vz;            // zero rows of other types
      c = __builtin_amdgcn_wmma_f32_16x16x32_f16(false, at, false, b, (short)0, c, false, false);
    }
  }
  _Float16* out = (p == 0) ? kf : (p == 1) ? qf : vf;
#pragma unroll
  for (int i = 0; i < 8; ++i) {
    int orow = tile * 16 + i + half * 8;
    out[(size_t)orow * DHID + col] = (_Float16)c[i];
  }
}

// ---------------- K4: edge attention scores + segment max ----------------
// gid = edge*8 + head
__global__ __launch_bounds__(256) void hgt_escore(
    const _Float16* __restrict__ kf, const _Float16* __restrict__ qf,
    const int* __restrict__ src, const int* __restrict__ dst, const int* __restrict__ etp,
    const float* __restrict__ ratt, const float* __restrict__ rpri,
    float* __restrict__ eb, unsigned* __restrict__ mb) {
  __shared__ _Float16 satt[TET * NH * DKH * DKH];          // 32 KB
  for (int i = threadIdx.x; i < TET * NH * DKH * DKH; i += 256) satt[i] = (_Float16)ratt[i];
  __syncthreads();
  int gid = blockIdx.x * 256 + threadIdx.x;
  if (gid >= NE * NH) return;
  int e = gid >> 3, h = gid & 7;
  int s = src[e], d = dst[e], et = etp[e];
  float kv[16], qv[16];
  {
    v8h k0 = *(const v8h*)(kf + (size_t)s * DHID + h * DKH);
    v8h k1 = *(const v8h*)(kf + (size_t)s * DHID + h * DKH + 8);
    v8h q0 = *(const v8h*)(qf + (size_t)d * DHID + h * DKH);
    v8h q1 = *(const v8h*)(qf + (size_t)d * DHID + h * DKH + 8);
#pragma unroll
    for (int i = 0; i < 8; ++i) {
      kv[i] = (float)k0[i]; kv[i + 8] = (float)k1[i];
      qv[i] = (float)q0[i]; qv[i + 8] = (float)q1[i];
    }
  }
  const _Float16* A = satt + (et * NH + h) * (DKH * DKH);
  float acc = 0.f;
#pragma unroll
  for (int j = 0; j < DKH; ++j) {                // e = sum_{i,j} k_i * Att[i][j] * q_j
    float tmp = 0.f;
#pragma unroll
    for (int i = 0; i < DKH; ++i) tmp += kv[i] * (float)A[i * DKH + j];
    acc += qv[j] * tmp;
  }
  float ev = acc * rpri[et * NH + h] * 0.25f;    // scale = (DH/H)^-0.5 = 1/4
  eb[gid] = ev;
  atomicMax(&mb[d * NH + h], fkey(ev));
}

// ---------------- K5: exp(e - max) + segment sum ----------------
__global__ __launch_bounds__(256) void hgt_exp(
    const int* __restrict__ dst, float* __restrict__ eb,
    const unsigned* __restrict__ mb, float* __restrict__ db) {
  int gid = blockIdx.x * 256 + threadIdx.x;
  if (gid >= NE * NH) return;
  int e = gid >> 3, h = gid & 7;
  int d = dst[e];
  float m = fdekey(mb[d * NH + h]);
  float ex = __expf(eb[gid] - m);
  eb[gid] = ex;
  atomicAdd(&db[d * NH + h], ex);
}

// ---------------- K6: weighted message aggregation ----------------
__global__ __launch_bounds__(256) void hgt_agg(
    const _Float16* __restrict__ vf,
    const int* __restrict__ src, const int* __restrict__ dst, const int* __restrict__ etp,
    const float* __restrict__ rmsg, const float* __restrict__ eb,
    const float* __restrict__ db, float* __restrict__ aggf) {
  __shared__ _Float16 smsg[TET * NH * DKH * DKH];          // 32 KB
  for (int i = threadIdx.x; i < TET * NH * DKH * DKH; i += 256) smsg[i] = (_Float16)rmsg[i];
  __syncthreads();
  int gid = blockIdx.x * 256 + threadIdx.x;
  if (gid >= NE * NH) return;
  int e = gid >> 3, h = gid & 7;
  int s = src[e], d = dst[e], et = etp[e];
  float a = eb[gid] / db[d * NH + h];
  float vv[16];
  {
    v8h v0 = *(const v8h*)(vf + (size_t)s * DHID + h * DKH);
    v8h v1 = *(const v8h*)(vf + (size_t)s * DHID + h * DKH + 8);
#pragma unroll
    for (int i = 0; i < 8; ++i) { vv[i] = (float)v0[i]; vv[i + 8] = (float)v1[i]; }
  }
  const _Float16* M = smsg + (et * NH + h) * (DKH * DKH);
  float* outp = aggf + (size_t)d * DHID + h * DKH;
#pragma unroll
  for (int j = 0; j < DKH; ++j) {                // vt_j = sum_i v_i * Msg[i][j]
    float tmp = 0.f;
#pragma unroll
    for (int i = 0; i < DKH; ++i) tmp += vv[i] * (float)M[i * DKH + j];
    atomicAdd(&outp[j], a * tmp);
  }
}

// ---------------- K7: agg f32 -> f16 ----------------
__global__ __launch_bounds__(256) void hgt_cvt_agg(const float* aggf, _Float16* aggh) {
  int idx = blockIdx.x * 256 + threadIdx.x;
  if (idx < NN * DHID) aggh[idx] = (_Float16)aggf[idx];
}

// ---------------- K8: node update: a-proj (masked WMMA) + self-loop + bias + relu -----
__global__ __launch_bounds__(256) void hgt_out(
    const _Float16* __restrict__ aggh, const _Float16* __restrict__ nodef,
    const int* __restrict__ ntype, const _Float16* __restrict__ wt,
    const _Float16* __restrict__ loopwt, const float* __restrict__ bias,
    float* __restrict__ dout) {
  const int tile = blockIdx.x;
  const int wave = threadIdx.x >> 5;
  const int lane = threadIdx.x & 31;
  const int half = lane >> 4, lrow = lane & 15;
  const int rowA = tile * 16 + lrow;
  const int ty   = ntype[rowA];
  const int col  = wave * 16 + lrow;
  const _Float16* wbase = wt + (size_t)3 * (TNT * DHID * DIN);   // 'a' projection weights
  const v16h vz = {};
  v8f c = {};
#pragma unroll
  for (int kk = 0; kk < 4; ++kk) {
    union { v16h v; v8h h[2]; } aA, aN;
    const _Float16* pg = aggh  + (size_t)rowA * DHID + kk * 32 + half * 16;
    const _Float16* pn = nodef + (size_t)rowA * DIN  + kk * 32 + half * 16;
    aA.h[0] = *(const v8h*)(pg);      aA.h[1] = *(const v8h*)(pg + 8);
    aN.h[0] = *(const v8h*)(pn);      aN.h[1] = *(const v8h*)(pn + 8);
#pragma unroll
    for (int t = 0; t < TNT; ++t) {
      v16h b = *(const v16h*)(wbase + ((size_t)t * DHID + col) * DIN + kk * 32 + half * 16);
      v16h at = (ty == t) ? aA.v : vz;
      c = __builtin_amdgcn_wmma_f32_16x16x32_f16(false, at, false, b, (short)0, c, false, false);
    }
    v16h bl = *(const v16h*)(loopwt + (size_t)col * DIN + kk * 32 + half * 16);
    c = __builtin_amdgcn_wmma_f32_16x16x32_f16(false, aN.v, false, bl, (short)0, c, false, false);
  }
  float bv = bias[col];
#pragma unroll
  for (int i = 0; i < 8; ++i) {
    int orow = tile * 16 + i + half * 8;
    float r = c[i] + bv;
    dout[(size_t)orow * DHID + col] = r > 0.f ? r : 0.f;
  }
}

// ---------------- host launcher ----------------
extern "C" void kernel_launch(void* const* d_in, const int* in_sizes, int n_in,
                              void* d_out, int out_size, void* d_ws, size_t ws_size,
                              hipStream_t stream) {
  const float* node_feat = (const float*)d_in[0];
  const int*   node_type = (const int*)d_in[1];
  const int*   src       = (const int*)d_in[2];
  const int*   dst       = (const int*)d_in[3];
  const int*   etp       = (const int*)d_in[4];
  const float* ck = (const float*)d_in[5],  *bk = (const float*)d_in[6];
  const float* cq = (const float*)d_in[7],  *bq = (const float*)d_in[8];
  const float* cv = (const float*)d_in[9],  *bv = (const float*)d_in[10];
  const float* ca = (const float*)d_in[11], *ba = (const float*)d_in[12];
  const float* rpri  = (const float*)d_in[13];
  const float* ratt  = (const float*)d_in[14];
  const float* rmsg  = (const float*)d_in[15];
  const float* loopw = (const float*)d_in[16];
  const float* bias  = (const float*)d_in[17];
  float* dout = (float*)d_out;

  char* ws = (char*)d_ws;
  _Float16* wt     = (_Float16*)(ws + OFF_WT);
  _Float16* loopwt = (_Float16*)(ws + OFF_LOOPW);
  _Float16* nodef  = (_Float16*)(ws + OFF_NODEF);
  _Float16* kf     = (_Float16*)(ws + OFF_KF);
  _Float16* qf     = (_Float16*)(ws + OFF_QF);
  _Float16* vf     = (_Float16*)(ws + OFF_VF);
  _Float16* aggh   = (_Float16*)(ws + OFF_AGGH);
  float*    eb     = (float*)(ws + OFF_EB);
  unsigned* mb     = (unsigned*)(ws + OFF_MB);
  float*    db     = (float*)(ws + OFF_DB);
  float*    aggf   = (float*)(ws + OFF_AGGF);

  const int NTILES = NN / 16;                      // 3125 (N divisible by 16)
  const int G_W    = (4 * TNT * DHID * DIN + DHID * DIN + 255) / 256;
  const int G_N    = (NN * DIN + 255) / 256;       // 25000
  const int G_E    = (NE * NH + 255) / 256;        // 25000

  hgt_prep_w  <<<G_W, 256, 0, stream>>>(ck, bk, cq, bq, cv, bv, ca, ba, loopw, wt, loopwt);
  hgt_cvt_nodes<<<G_N, 256, 0, stream>>>(node_feat, nodef);
  hgt_init    <<<G_N, 256, 0, stream>>>(aggf, mb, db);
  hgt_proj    <<<dim3(NTILES, 3), 256, 0, stream>>>(nodef, node_type, wt, kf, qf, vf);
  hgt_escore  <<<G_E, 256, 0, stream>>>(kf, qf, src, dst, etp, ratt, rpri, eb, mb);
  hgt_exp     <<<G_E, 256, 0, stream>>>(dst, eb, mb, db);
  hgt_agg     <<<G_E, 256, 0, stream>>>(vf, src, dst, etp, rmsg, eb, db, aggf);
  hgt_cvt_agg <<<G_N, 256, 0, stream>>>(aggf, aggh);
  hgt_out     <<<NTILES, 256, 0, stream>>>(aggh, nodef, node_type, wt, loopwt, bias, dout);
}
